// LSTM_DeepPredictor_54941221650542
// MI455X (gfx1250) — compile-verified, matrix operand order checked by reading, not run on hardware
//
#include <hip/hip_runtime.h>

// Problem constants (match reference)
#define BB   256
#define TT   256
#define FF   64
#define HH   512
#define LL   4
#define G4H  (4 * HH)   // 2048
#define NWG  128        // persistent grid: (HH/32) * (BB/32)

// ---------------------------------------------------------------------------
// bf16 helpers (stored as ushort in workspace)
// ---------------------------------------------------------------------------
static __device__ __forceinline__ unsigned short f2bf(float f) {
    unsigned u = __float_as_uint(f);
    unsigned r = (u + 0x7FFFu + ((u >> 16) & 1u)) >> 16;   // round-to-nearest-even
    return (unsigned short)r;
}
static __device__ __forceinline__ float bf2f(unsigned short b) {
    return __uint_as_float(((unsigned)b) << 16);
}

// ---------------------------------------------------------------------------
// WMMA fragment types (CDNA5 wave32, 16x16x32 bf16 -> f32)
// ---------------------------------------------------------------------------
typedef __attribute__((ext_vector_type(16))) __bf16 v16bf;
typedef __attribute__((ext_vector_type(8)))  float  v8f;

union FragU {
    v16bf v;
    uint4 q[2];
};

// 16-bit A/B fragment from a row-major matrix (ld in elements).
// Lane ln holds row (ln&15); halves 0..7 = K k0+((ln>>4)<<3)..+7, halves
// 8..15 = same +16 (per CDNA5 ISA 16-bit matrix VGPR layout).
static __device__ __forceinline__ v16bf load_frag_g(const unsigned short* p,
                                                    int row0, int k0, int ld) {
    unsigned ln  = threadIdx.x & 31u;
    int row = row0 + (int)(ln & 15u);
    int kb  = k0 + (int)((ln >> 4) << 3);
    const unsigned short* base = p + (size_t)row * (size_t)ld + kb;
    FragU u;
    u.q[0] = *(const uint4*)(base);
    u.q[1] = *(const uint4*)(base + 16);
    return u.v;
}

// Same fragment gather but from an LDS-resident row-major slice.
static __device__ __forceinline__ v16bf load_frag_lds(const unsigned short* rowbase,
                                                      int k0, int ld) {
    unsigned ln = threadIdx.x & 31u;
    const unsigned short* base =
        rowbase + (size_t)(ln & 15u) * (size_t)ld + k0 + (int)((ln >> 4) << 3);
    FragU u;
    u.q[0] = *(const uint4*)(base);
    u.q[1] = *(const uint4*)(base + 16);
    return u.v;
}

static __device__ __forceinline__ v8f wmma_bf16(v16bf a, v16bf b, v8f c) {
    return __builtin_amdgcn_wmma_f32_16x16x32_bf16(
        /*neg_a=*/false, a, /*neg_b=*/false, b,
        /*c_mod=*/(short)0, c, /*reuse_a=*/false, /*reuse_b=*/false);
}

// ---------------------------------------------------------------------------
// Software-pipelined GEMM segment: D += A_seg x B_seg over nkt k-tiles.
// Depth-1 double buffering: loads for k+1 are issued before WMMA of k, so
// the scheduler can hold a load batch in flight across every WMMA pair
// instead of draining LOADcnt/DScnt to zero each step.
// ---------------------------------------------------------------------------
static __device__ __forceinline__ void gemm_seg(
        const unsigned short* __restrict__ asrc, int arow, int ald,
        const unsigned short* __restrict__ wrow0,
        const unsigned short* __restrict__ wrow1,
        int koff, int nkt, int ldw, v8f& acc0, v8f& acc1)
{
    v16bf a  = load_frag_g(asrc, arow, 0, ald);
    v16bf b0 = load_frag_lds(wrow0, koff, ldw);
    v16bf b1 = load_frag_lds(wrow1, koff, ldw);
    for (int kt = 0; kt + 1 < nkt; ++kt) {
        v16bf an  = load_frag_g(asrc, arow, (kt + 1) * 32, ald);
        v16bf b0n = load_frag_lds(wrow0, koff + (kt + 1) * 32, ldw);
        v16bf b1n = load_frag_lds(wrow1, koff + (kt + 1) * 32, ldw);
        acc0 = wmma_bf16(a, b0, acc0);
        acc1 = wmma_bf16(a, b1, acc1);
        a = an; b0 = b0n; b1 = b1n;
    }
    acc0 = wmma_bf16(a, b0, acc0);
    acc1 = wmma_bf16(a, b1, acc1);
}

// ---------------------------------------------------------------------------
// Device-wide barrier: monotonic counter, release on arrive / acquire on spin.
// ---------------------------------------------------------------------------
static __device__ __forceinline__ void grid_sync(unsigned* counter, unsigned target) {
    __threadfence();            // make this WG's stores visible at device scope
    __syncthreads();
    if (threadIdx.x == 0) {
        __hip_atomic_fetch_add(counter, 1u, __ATOMIC_ACQ_REL,
                               __HIP_MEMORY_SCOPE_AGENT);
        while (__hip_atomic_load(counter, __ATOMIC_ACQUIRE,
                                 __HIP_MEMORY_SCOPE_AGENT) < target) {
            __builtin_amdgcn_s_sleep(8);
        }
    }
    __syncthreads();
}

// ---------------------------------------------------------------------------
// Prep kernels
// ---------------------------------------------------------------------------

// Wcat[n][k] = (k < in_dim) ? Wih[n][k] : Whh[n][k-in_dim], converted to bf16.
__global__ void cat_weights_kernel(const float* __restrict__ Wih,
                                   const float* __restrict__ Whh,
                                   unsigned short* __restrict__ Wcat,
                                   int in_dim) {
    const int ldw = in_dim + HH;
    const long total = (long)G4H * (long)ldw;
    for (long i = (long)blockIdx.x * blockDim.x + threadIdx.x; i < total;
         i += (long)gridDim.x * blockDim.x) {
        long n = i / ldw;
        int  k = (int)(i - n * ldw);
        float v = (k < in_dim) ? Wih[n * in_dim + k] : Whh[n * HH + (k - in_dim)];
        Wcat[i] = f2bf(v);
    }
}

__global__ void add_bias_kernel(const float* __restrict__ bih,
                                const float* __restrict__ bhh,
                                float* __restrict__ bsum,
                                unsigned* __restrict__ sync) {
    if (blockIdx.x == 0 && threadIdx.x == 0) *sync = 0u;   // reset grid barrier
    for (int i = blockIdx.x * blockDim.x + threadIdx.x; i < LL * G4H;
         i += gridDim.x * blockDim.x)
        bsum[i] = bih[i] + bhh[i];
}

// x [B][T][F] fp32 -> xbf [T][B][F] bf16 (time-major for the scan)
__global__ void prep_x_kernel(const float* __restrict__ x,
                              unsigned short* __restrict__ xbf) {
    const long total = (long)BB * TT * FF;
    for (long i = (long)blockIdx.x * blockDim.x + threadIdx.x; i < total;
         i += (long)gridDim.x * blockDim.x) {
        int  f  = (int)(i % FF);
        long bt = i / FF;
        int  t  = (int)(bt % TT);
        long b  = bt / TT;
        xbf[((long)t * BB + b) * FF + f] = f2bf(x[i]);
    }
}

// ---------------------------------------------------------------------------
// Persistent per-layer LSTM scan.
// Grid: NWG = (HH/32 col groups) x (BB/32 row groups); block: 256 thr = 8 waves.
// Wave w = (gate g = w&3, batch subtile ms = w>>2). Each WG owns a fixed
// 32(batch) x 32(hidden) tile: weights staged once in LDS, c-state in regs.
// ---------------------------------------------------------------------------
__global__ __launch_bounds__(256) void lstm_layer_kernel(
        const unsigned short* __restrict__ in_seq, // [T][B][in_dim] bf16
        int in_dim,
        const unsigned short* __restrict__ Wcat,   // [4H][in_dim+HH] bf16
        const float* __restrict__ bsum,            // [4H]
        unsigned short* __restrict__ ys,           // [T][B][HH] bf16 out
        unsigned* __restrict__ sync,
        unsigned goal0)                            // layer * TT * NWG
{
    const int ldw = in_dim + HH;
    extern __shared__ __align__(16) unsigned short Wlds[];   // [4*32][ldw]
    __shared__ float zs[4][32][32];

    const int tid  = threadIdx.x;
    const int wave = tid >> 5;
    const int g    = wave & 3;          // gate: 0=i 1=f 2=g 3=o
    const int ms   = wave >> 2;         // batch subtile (0/1)
    const unsigned ln = tid & 31u;

    const int nb = blockIdx.x & 15;     // HH/32 = 16 column groups
    const int mb = blockIdx.x >> 4;     // BB/32 = 8  row groups
    const int n0 = nb * 32;
    const int m0 = mb * 32;

    // ---- stage this WG's weight slice into LDS (once per layer) ----
    {
        const int vecPerRow = ldw / 8;              // b128 chunks per row
        const int total = 128 * vecPerRow;          // 4 gates * 32 cols
        for (int i = tid; i < total; i += 256) {
            int r  = i / vecPerRow;
            int v  = i - r * vecPerRow;
            int gg = r >> 5, nl = r & 31;
            const uint4* src =
                (const uint4*)(Wcat + (size_t)(gg * HH + n0 + nl) * ldw) + v;
            *((uint4*)(&Wlds[(size_t)r * ldw]) + v) = *src;
        }
    }
    __syncthreads();

    // per-wave bias registers (C/D tile N = lane&15)
    const float bia0 = bsum[g * HH + n0 +      (int)(ln & 15u)];
    const float bia1 = bsum[g * HH + n0 + 16 + (int)(ln & 15u)];
    const unsigned short* wrow0 = &Wlds[(size_t)(g * 32)      * ldw];
    const unsigned short* wrow1 = &Wlds[(size_t)(g * 32 + 16) * ldw];
    const int arow = m0 + ms * 16;

    // c-state registers: element e = tid + j*256 of the 32x32 tile
    float creg[4] = {0.0f, 0.0f, 0.0f, 0.0f};

    for (int t = 0; t < TT; ++t) {
        const unsigned short* xin = in_seq + (size_t)t * BB * in_dim;
        v8f acc0 = {};
        v8f acc1 = {};

        // input projection: K = [0, in_dim)
        gemm_seg(xin, arow, in_dim, wrow0, wrow1, 0, in_dim / 32, ldw,
                 acc0, acc1);
        // recurrent part: K = [in_dim, in_dim+HH)
        if (t > 0) {
            const unsigned short* hprev = ys + (size_t)(t - 1) * BB * HH;
            gemm_seg(hprev, arow, HH, wrow0, wrow1, in_dim, HH / 32, ldw,
                     acc0, acc1);
        }

        // gate pre-activations -> LDS. C/D layout: VGPR r, lanes 0-15 M=r,
        // lanes 16-31 M=8+r; N = lane&15.
        {
            const int mh = (int)((ln >> 4) << 3);
            const int nl0 = (int)(ln & 15u);
            #pragma unroll
            for (int r = 0; r < 8; ++r) {
                zs[g][ms * 16 + r + mh][nl0]      = acc0[r] + bia0;
                zs[g][ms * 16 + r + mh][16 + nl0] = acc1[r] + bia1;
            }
        }
        __syncthreads();

        // fused gates + state update: 1024 elems / 256 threads = 4 each
        unsigned short* hout = ys + (size_t)t * BB * HH;
        #pragma unroll
        for (int j = 0; j < 4; ++j) {
            int e  = tid + j * 256;
            int ml = e >> 5;
            int nl = e & 31;
            float zi = zs[0][ml][nl];
            float zf = zs[1][ml][nl];
            float zg = zs[2][ml][nl];
            float zo = zs[3][ml][nl];
            float si = 1.0f / (1.0f + __expf(-zi));
            float sf = 1.0f / (1.0f + __expf(-zf));
            float tg = tanhf(zg);
            float so = 1.0f / (1.0f + __expf(-zo));
            float cn = sf * creg[j] + si * tg;
            creg[j] = cn;
            hout[(size_t)(m0 + ml) * HH + (n0 + nl)] = f2bf(so * tanhf(cn));
        }

        // device-wide step barrier (also fences zs reuse via its syncthreads)
        grid_sync(sync, goal0 + (unsigned)(t + 1) * NWG);
    }
}

// ---------------------------------------------------------------------------
// Head: out[b] = relu(dot(h_last[b], Wlin) + blin)
// ---------------------------------------------------------------------------
__global__ void head_kernel(const unsigned short* __restrict__ hlast,
                            const float* __restrict__ Wlin,
                            const float* __restrict__ blin,
                            float* __restrict__ out) {
    const int b = blockIdx.x;
    const int tid = threadIdx.x;           // 64 threads
    __shared__ float red[64];
    float s = 0.0f;
    for (int k = tid; k < HH; k += 64)
        s += bf2f(hlast[(long)b * HH + k]) * Wlin[k];
    red[tid] = s;
    __syncthreads();
    for (int off = 32; off > 0; off >>= 1) {
        if (tid < off) red[tid] += red[tid + off];
        __syncthreads();
    }
    if (tid == 0) out[b] = fmaxf(red[0] + blin[0], 0.0f);
}

// ---------------------------------------------------------------------------
// Host launcher
// ---------------------------------------------------------------------------
extern "C" void kernel_launch(void* const* d_in, const int* in_sizes, int n_in,
                              void* d_out, int out_size, void* d_ws, size_t ws_size,
                              hipStream_t stream) {
    (void)in_sizes; (void)n_in; (void)out_size; (void)ws_size;

    const float* x    = (const float*)d_in[0];   // [B,T,F]
    const float* Wih0 = (const float*)d_in[1];   // [4H,F]
    const float* Wihr = (const float*)d_in[2];   // [L-1,4H,H]
    const float* Whh  = (const float*)d_in[3];   // [L,4H,H]
    const float* bih  = (const float*)d_in[4];   // [L,4H]
    const float* bhh  = (const float*)d_in[5];   // [L,4H]
    const float* Wlin = (const float*)d_in[6];   // [1,H]
    const float* blin = (const float*)d_in[7];   // [1]
    float* out = (float*)d_out;

    // ---- workspace carving (256B aligned) ----
    char*  ws  = (char*)d_ws;
    size_t off = 0;
    auto carve = [&](size_t bytes) -> char* {
        char* r = ws + off;
        off = (off + bytes + 255) & ~(size_t)255;
        return r;
    };
    unsigned short* Wcat[LL];
    Wcat[0] = (unsigned short*)carve((size_t)G4H * (FF + HH) * 2);
    for (int l = 1; l < LL; ++l)
        Wcat[l] = (unsigned short*)carve((size_t)G4H * (HH + HH) * 2);
    float*          bsum = (float*)carve((size_t)LL * G4H * 4);
    unsigned short* xbf  = (unsigned short*)carve((size_t)TT * BB * FF * 2);
    unsigned short* ysA  = (unsigned short*)carve((size_t)TT * BB * HH * 2);
    unsigned short* ysB  = (unsigned short*)carve((size_t)TT * BB * HH * 2);
    unsigned*       sync = (unsigned*)carve(256);

    // allow large dynamic LDS for the persistent kernel (idempotent; errors ignored)
    (void)hipFuncSetAttribute((const void*)lstm_layer_kernel,
                              hipFuncAttributeMaxDynamicSharedMemorySize,
                              4 * 32 * (HH + HH) * 2);

    // ---- one-time prep (runs every call; deterministic) ----
    cat_weights_kernel<<<256, 256, 0, stream>>>(Wih0, Whh, Wcat[0], FF);
    for (int l = 1; l < LL; ++l)
        cat_weights_kernel<<<256, 256, 0, stream>>>(
            Wihr + (size_t)(l - 1) * G4H * HH, Whh + (size_t)l * G4H * HH,
            Wcat[l], HH);
    add_bias_kernel<<<32, 256, 0, stream>>>(bih, bhh, bsum, sync);
    prep_x_kernel<<<1024, 256, 0, stream>>>(x, xbf);

    // ---- one persistent launch per layer ----
    const unsigned short* in_seq = xbf;
    int in_dim = FF;
    unsigned short* ys_out = ysA;
    for (int l = 0; l < LL; ++l) {
        const size_t dynLds = (size_t)4 * 32 * (in_dim + HH) * 2;
        lstm_layer_kernel<<<NWG, 256, dynLds, stream>>>(
            in_seq, in_dim, Wcat[l], bsum + l * G4H, ys_out, sync,
            (unsigned)l * TT * NWG);
        in_seq = ys_out;
        in_dim = HH;
        ys_out = (ys_out == ysA) ? ysB : ysA;
    }

    // ---- linear + relu head on last timestep of last layer ----
    head_kernel<<<BB, 64, 0, stream>>>(
        in_seq + (size_t)(TT - 1) * BB * HH, Wlin, blin, out);
}
